// TemporalDiffusionTransformerDecoderLayer_52175262711851
// MI455X (gfx1250) — compile-verified, matrix operand order checked by reading, not run on hardware
//
#include <hip/hip_runtime.h>
#include <hip/hip_bf16.h>
#include <math.h>

typedef unsigned short u16;
typedef __attribute__((ext_vector_type(8)))  u16    u16x8;
typedef __attribute__((ext_vector_type(16))) u16    u16x16;
typedef __attribute__((ext_vector_type(16))) __bf16 bf16x16;
typedef __attribute__((ext_vector_type(8)))  float  f32x8;

#define BB    4
#define TT    1024
#define NC    256
#define DD    512
#define TE_   512
#define FF    2048
#define HH    8
#define DH_   64

__device__ __forceinline__ u16 f2bf(float f) {
  union { float f; unsigned u; } x; x.f = f;
  unsigned r = x.u + 0x7fffu + ((x.u >> 16) & 1u);
  return (u16)(r >> 16);
}

__device__ __forceinline__ f32x8 wmma_bf16(const u16x16& a, const u16x16& b, f32x8 c) {
  return __builtin_amdgcn_wmma_f32_16x16x32_bf16(
      false, __builtin_bit_cast(bf16x16, a),
      false, __builtin_bit_cast(bf16x16, b),
      (short)0, c, false, false);
}

// CDNA5 async global->LDS copy of 16 bytes (tracked by ASYNCcnt).
__device__ __forceinline__ void async_copy_b128(const u16* gsrc, u16* lds_dst) {
  unsigned lds_off = (unsigned)(unsigned long long)lds_dst;  // addr[31:0] = LDS byte offset
  unsigned long long ga = (unsigned long long)gsrc;
  asm volatile("global_load_async_to_lds_b128 %0, %1, off"
               :: "v"(lds_off), "v"(ga) : "memory");
}
__device__ __forceinline__ void wait_asynccnt0() {
  asm volatile("s_wait_asynccnt 0" ::: "memory");
}

// ---------------- f32 -> bf16 convert ----------------
__global__ void cvt_f32_bf16(const float* __restrict__ in, u16* __restrict__ out, int n) {
  int i = blockIdx.x * blockDim.x + threadIdx.x;
  if (i < n) out[i] = f2bf(in[i]);
}

// ---------------- emb: ss = silu(emb) @ W + b  (M=4, tiny -> VALU) ----------------
__global__ void emb_ss_kernel(const float* __restrict__ emb, const float* __restrict__ W,
                              const float* __restrict__ bias, float* __restrict__ ss,
                              int Brows, int K, int Nout) {
  int idx = blockIdx.x * blockDim.x + threadIdx.x;
  if (idx >= Brows * Nout) return;
  int b = idx / Nout, j = idx % Nout;
  float acc = bias[j];
  for (int k = 0; k < K; ++k) {
    float e = emb[(size_t)b * K + k];
    float s = e / (1.f + __expf(-e));
    acc += s * W[(size_t)k * Nout + j];
  }
  ss[idx] = acc;
}

// ---------------- LayerNorm -> bf16 ----------------
__global__ void __launch_bounds__(256) ln_kernel(const float* __restrict__ x,
                                                 const float* __restrict__ g,
                                                 const float* __restrict__ bta,
                                                 u16* __restrict__ out, int dim) {
  int row = blockIdx.x;
  int t = threadIdx.x;
  const float* xr = x + (size_t)row * dim;
  float s = 0.f, s2 = 0.f;
  for (int i = t; i < dim; i += 256) { float v = xr[i]; s += v; s2 += v * v; }
  __shared__ float red[256];
  red[t] = s; __syncthreads();
  for (int o = 128; o > 0; o >>= 1) { if (t < o) red[t] += red[t + o]; __syncthreads(); }
  float mean = red[0] / dim; __syncthreads();
  red[t] = s2; __syncthreads();
  for (int o = 128; o > 0; o >>= 1) { if (t < o) red[t] += red[t + o]; __syncthreads(); }
  float var = red[0] / dim - mean * mean;
  float rstd = rsqrtf(var + 1e-5f);
  for (int i = t; i < dim; i += 256)
    out[(size_t)row * dim + i] = f2bf((xr[i] - mean) * rstd * g[i] + bta[i]);
}

// ---------------- fused adaLN modulate + SiLU -> bf16 ----------------
__global__ void __launch_bounds__(256) styl_mod_kernel(const float* __restrict__ y,
                                                       const float* __restrict__ g,
                                                       const float* __restrict__ bta,
                                                       const float* __restrict__ ss,
                                                       u16* __restrict__ out, int dim, int T) {
  int row = blockIdx.x;
  int b = row / T;
  int t = threadIdx.x;
  const float* yr = y + (size_t)row * dim;
  float s = 0.f, s2 = 0.f;
  for (int i = t; i < dim; i += 256) { float v = yr[i]; s += v; s2 += v * v; }
  __shared__ float red[256];
  red[t] = s; __syncthreads();
  for (int o = 128; o > 0; o >>= 1) { if (t < o) red[t] += red[t + o]; __syncthreads(); }
  float mean = red[0] / dim; __syncthreads();
  red[t] = s2; __syncthreads();
  for (int o = 128; o > 0; o >>= 1) { if (t < o) red[t] += red[t + o]; __syncthreads(); }
  float var = red[0] / dim - mean * mean;
  float rstd = rsqrtf(var + 1e-5f);
  const float* sc = ss + (size_t)b * 2 * dim;
  for (int i = t; i < dim; i += 256) {
    float h = (yr[i] - mean) * rstd * g[i] + bta[i];
    h = h * (1.f + sc[i]) + sc[dim + i];
    float o2 = h / (1.f + __expf(-h));          // SiLU
    out[(size_t)row * dim + i] = f2bf(o2);
  }
}

// ---------------- bf16 WMMA GEMM: out = act(A@W + bias) (+res) ----------------
// A: M x K bf16 row-major.  W: K x N bf16 row-major.  Block tile 64(M) x 128(N).
// K staged in slabs of 64 (two 32-deep WMMA sub-steps per stage -> 8 WMMAs/wave/stage).
// A slab staged via async global->LDS (CDNA5 TDM-adjacent path); B slab staged
// transposed (Bt[n][k]) so B fragments are contiguous ds_load_b128 pairs.
__global__ void __launch_bounds__(256) gemm_wmma(const u16* __restrict__ A,
                                                 const u16* __restrict__ W,
                                                 const float* __restrict__ bias,
                                                 const float* __restrict__ res,
                                                 float* __restrict__ outF,
                                                 u16* __restrict__ outB,
                                                 int M, int N, int K, int act) {
  __shared__ __align__(16) u16 At[64][64];     // 8 KB
  __shared__ __align__(16) u16 Bt[128][64];    // 16 KB, transposed
  int tid  = threadIdx.x;
  int lane = tid & 31, w = tid >> 5;
  int wm = w & 1, wn = w >> 1;
  int m0 = blockIdx.x * 64, n0 = blockIdx.y * 128;
  int ln16 = lane & 15, hl = lane >> 4;

  f32x8 acc[2][2];
#pragma unroll
  for (int a = 0; a < 2; ++a)
#pragma unroll
    for (int b = 0; b < 2; ++b) acc[a][b] = {};

  for (int kc = 0; kc < K; kc += 64) {
    // stage A slab (64x64) via async global->LDS b128 copies (2 per thread)
#pragma unroll
    for (int rep = 0; rep < 2; ++rep) {
      int chunk = tid + rep * 256;            // 512 chunks of 8 u16
      int r = chunk >> 3, c8 = (chunk & 7) * 8;
      async_copy_b128(A + (size_t)(m0 + r) * K + kc + c8, &At[r][c8]);
    }
    // stage B slab transposed (64x128 -> Bt[128][64]) : 4 b128 loads per thread
#pragma unroll
    for (int rep = 0; rep < 4; ++rep) {
      int chunk = tid + rep * 256;            // 1024 chunks
      int kk = chunk >> 4, cg = (chunk & 15) * 8;
      u16x8 v = *(const u16x8*)(W + (size_t)(kc + kk) * N + n0 + cg);
#pragma unroll
      for (int j = 0; j < 8; ++j) Bt[cg + j][kk] = v[j];
      if (kc + 64 < K)
        __builtin_prefetch(W + (size_t)(kc + 64 + kk) * N + n0 + cg, 0, 0);
    }
    wait_asynccnt0();
    __syncthreads();

#pragma unroll
    for (int ks = 0; ks < 2; ++ks) {          // two 32-deep K sub-steps
      int kb = ks * 32;
      u16x16 af[2], bfr[2];
#pragma unroll
      for (int ms = 0; ms < 2; ++ms) {
        int r = wm * 32 + ms * 16 + ln16;
        u16x8 lo = *(const u16x8*)(&At[r][kb + hl * 8]);
        u16x8 hi = *(const u16x8*)(&At[r][kb + 16 + hl * 8]);
#pragma unroll
        for (int j = 0; j < 8; ++j) { af[ms][j] = lo[j]; af[ms][8 + j] = hi[j]; }
      }
#pragma unroll
      for (int ns = 0; ns < 2; ++ns) {
        int n = wn * 32 + ns * 16 + ln16;
        u16x8 lo = *(const u16x8*)(&Bt[n][kb + hl * 8]);
        u16x8 hi = *(const u16x8*)(&Bt[n][kb + 16 + hl * 8]);
#pragma unroll
        for (int j = 0; j < 8; ++j) { bfr[ns][j] = lo[j]; bfr[ns][8 + j] = hi[j]; }
      }
#pragma unroll
      for (int ms = 0; ms < 2; ++ms)
#pragma unroll
        for (int ns = 0; ns < 2; ++ns)
          acc[ms][ns] = wmma_bf16(af[ms], bfr[ns], acc[ms][ns]);
    }
    __syncthreads();
  }

  // epilogue
#pragma unroll
  for (int ms = 0; ms < 2; ++ms)
#pragma unroll
    for (int ns = 0; ns < 2; ++ns) {
      int gn = n0 + wn * 32 + ns * 16 + ln16;
      float bv = bias[gn];
#pragma unroll
      for (int r = 0; r < 8; ++r) {
        int gm = m0 + wm * 32 + ms * 16 + r + hl * 8;
        float v = acc[ms][ns][r] + bv;
        if (act == 1)      v = v / (1.f + __expf(-v));                       // SiLU
        else if (act == 2) v = 0.5f * v * (1.f + erff(v * 0.70710678f));     // GELU exact
        size_t o = (size_t)gm * N + gn;
        if (res)  v += res[o];
        if (outF) outF[o] = v;
        if (outB) outB[o] = f2bf(v);
      }
    }
}

// ---------------- flash attention (bf16 QK/PV via WMMA, f32 softmax) ----------------
// q: [B,Tq,DD] bf16, k/v: [B,Tk,DD] bf16, head h at col h*DH_.  out f32 [B,Tq,DD].
// block = 4 waves; wave owns 16 queries; key blocks of 32.
__global__ void __launch_bounds__(128) attn_kernel(const u16* __restrict__ q,
                                                   const u16* __restrict__ k,
                                                   const u16* __restrict__ v,
                                                   float* __restrict__ out,
                                                   int Tq, int Tk) {
  __shared__ __align__(16) float Sld[4][16][32];
  __shared__ __align__(16) u16   Pld[4][16][32];
  __shared__ float Arow[4][16];
  const float scale = 0.125f;  // 1/sqrt(64)
  int tid = threadIdx.x, lane = tid & 31, w = tid >> 5;
  int ln16 = lane & 15, hl = lane >> 4;
  int qblocks = Tq / 64;
  int qblk = blockIdx.x % qblocks;
  int tmp  = blockIdx.x / qblocks;
  int h = tmp % HH, b = tmp / HH;
  int q0 = qblk * 64 + w * 16;
  const u16* qp = q + ((size_t)b * Tq + q0) * DD + h * DH_;
  const u16* kp = k + (size_t)b * Tk * DD + h * DH_;
  const u16* vp = v + (size_t)b * Tk * DD + h * DH_;

  // Q fragments (dh=64 -> two K-chunks of 32)
  u16x16 aq[2];
#pragma unroll
  for (int c = 0; c < 2; ++c) {
    const u16* basep = qp + (size_t)ln16 * DD + c * 32;
    u16x8 lo = *(const u16x8*)(basep + hl * 8);
    u16x8 hi = *(const u16x8*)(basep + 16 + hl * 8);
#pragma unroll
    for (int j = 0; j < 8; ++j) { aq[c][j] = lo[j]; aq[c][8 + j] = hi[j]; }
  }

  f32x8 o[4];
#pragma unroll
  for (int i = 0; i < 4; ++i) o[i] = {};
  float mi = -1e30f, li = 0.f;

  for (int k0 = 0; k0 < Tk; k0 += 32) {
    // S = scale * Q K^T for this 16x32 key block (two 16x16 N-tiles)
#pragma unroll
    for (int nt = 0; nt < 2; ++nt) {
      f32x8 s = {};
#pragma unroll
      for (int c = 0; c < 2; ++c) {
        const u16* basep = kp + (size_t)(k0 + nt * 16 + ln16) * DD + c * 32;
        u16x8 lo = *(const u16x8*)(basep + hl * 8);
        u16x8 hi = *(const u16x8*)(basep + 16 + hl * 8);
        u16x16 bf;
#pragma unroll
        for (int j = 0; j < 8; ++j) { bf[j] = lo[j]; bf[8 + j] = hi[j]; }
        s = wmma_bf16(aq[c], bf, s);
      }
#pragma unroll
      for (int r = 0; r < 8; ++r)
        Sld[w][r + hl * 8][nt * 16 + ln16] = s[r] * scale;
    }
    __syncthreads();

    // online softmax: lanes 0..15 each own one query row
    if (lane < 16) {
      float mx = mi;
      for (int j = 0; j < 32; ++j) mx = fmaxf(mx, Sld[w][lane][j]);
      float alpha = __expf(mi - mx);
      float sum = 0.f;
      for (int j = 0; j < 32; ++j) {
        float p = __expf(Sld[w][lane][j] - mx);
        Pld[w][lane][j] = f2bf(p);
        sum += p;
      }
      li = li * alpha + sum;
      mi = mx;
      Arow[w][lane] = alpha;
    }
    __syncthreads();

    // rescale O and accumulate P @ V
    float al[8];
#pragma unroll
    for (int r = 0; r < 8; ++r) al[r] = Arow[w][r + hl * 8];
#pragma unroll
    for (int nd = 0; nd < 4; ++nd)
#pragma unroll
      for (int r = 0; r < 8; ++r) o[nd][r] = o[nd][r] * al[r];

    u16x16 pf;
    {
      u16x8 lo = *(const u16x8*)(&Pld[w][ln16][hl * 8]);
      u16x8 hi = *(const u16x8*)(&Pld[w][ln16][16 + hl * 8]);
#pragma unroll
      for (int j = 0; j < 8; ++j) { pf[j] = lo[j]; pf[8 + j] = hi[j]; }
    }
#pragma unroll
    for (int nd = 0; nd < 4; ++nd) {
      u16x16 vf;
      int dcol = nd * 16 + ln16;
#pragma unroll
      for (int j = 0; j < 8; ++j) {
        vf[j]     = vp[(size_t)(k0 + hl * 8 + j) * DD + dcol];
        vf[8 + j] = vp[(size_t)(k0 + 16 + hl * 8 + j) * DD + dcol];
      }
      o[nd] = wmma_bf16(pf, vf, o[nd]);
    }
    __syncthreads();
  }

  if (lane < 16) Arow[w][lane] = (li > 0.f) ? (1.f / li) : 0.f;
  __syncthreads();
  float inv[8];
#pragma unroll
  for (int r = 0; r < 8; ++r) inv[r] = Arow[w][r + hl * 8];
#pragma unroll
  for (int nd = 0; nd < 4; ++nd)
#pragma unroll
    for (int r = 0; r < 8; ++r)
      out[((size_t)b * Tq + q0 + r + hl * 8) * DD + h * DH_ + nd * 16 + ln16] =
          o[nd][r] * inv[r];
}

// ---------------- host orchestration ----------------
extern "C" void kernel_launch(void* const* d_in, const int* in_sizes, int n_in,
                              void* d_out, int out_size, void* d_ws, size_t ws_size,
                              hipStream_t stream) {
  (void)in_sizes; (void)n_in; (void)out_size; (void)ws_size;
  const float* x    = (const float*)d_in[0];
  const float* xf   = (const float*)d_in[1];
  const float* embp = (const float*)d_in[2];
  auto F = [&](int i) { return (const float*)d_in[i]; };

  char* basep = (char*)d_ws;
  size_t off = 0;
  auto carve = [&](size_t bytes) -> void* {
    void* p = basep + off;
    off += (bytes + 255) & ~(size_t)255;
    return p;
  };
  const size_t W_DD = (size_t)DD * DD, W_F = (size_t)DD * FF;
  u16* wq_sa  = (u16*)carve(W_DD * 2);
  u16* wk_sa  = (u16*)carve(W_DD * 2);
  u16* wv_sa  = (u16*)carve(W_DD * 2);
  u16* wo_sa  = (u16*)carve(W_DD * 2);
  u16* wq_ca  = (u16*)carve(W_DD * 2);
  u16* wk_ca  = (u16*)carve(W_DD * 2);
  u16* wv_ca  = (u16*)carve(W_DD * 2);
  u16* wo_ca  = (u16*)carve(W_DD * 2);
  u16* wl1    = (u16*)carve(W_F * 2);
  u16* wl2    = (u16*)carve(W_F * 2);
  u16* wo_ff  = (u16*)carve(W_DD * 2);
  const size_t BT = (size_t)BB * TT, BN = (size_t)BB * NC;
  u16*   xn  = (u16*)carve(BT * DD * 2);
  u16*   cn  = (u16*)carve(BN * DD * 2);
  u16*   qb  = (u16*)carve(BT * DD * 2);
  u16*   kb  = (u16*)carve(BT * DD * 2);
  u16*   vb  = (u16*)carve(BT * DD * 2);
  float* yb  = (float*)carve(BT * DD * 4);
  u16*   hb  = (u16*)carve(BT * DD * 2);
  float* x1  = (float*)carve(BT * DD * 4);
  float* x2  = (float*)carve(BT * DD * 4);
  u16*   x2b = (u16*)carve(BT * DD * 2);
  u16*   h1  = (u16*)carve(BT * FF * 2);
  float* ss_sa = (float*)carve((size_t)BB * 2 * DD * 4);
  float* ss_ca = (float*)carve((size_t)BB * 2 * DD * 4);
  float* ss_ff = (float*)carve((size_t)BB * 2 * DD * 4);

  auto cvt = [&](int i, u16* dst, size_t n) {
    cvt_f32_bf16<<<(unsigned)((n + 255) / 256), 256, 0, stream>>>(F(i), dst, (int)n);
  };
  cvt(5, wq_sa, W_DD);  cvt(7, wk_sa, W_DD);  cvt(9, wv_sa, W_DD);
  cvt(15, wo_sa, W_DD); cvt(21, wq_ca, W_DD); cvt(23, wk_ca, W_DD);
  cvt(25, wv_ca, W_DD); cvt(31, wo_ca, W_DD);
  cvt(33, wl1, W_F);    cvt(35, wl2, W_F);    cvt(41, wo_ff, W_DD);

  // styl emb projections (B=4 rows -> VALU kernel)
  emb_ss_kernel<<<(BB * 2 * DD + 255) / 256, 256, 0, stream>>>(embp, F(11), F(12), ss_sa, BB, TE_, 2 * DD);
  emb_ss_kernel<<<(BB * 2 * DD + 255) / 256, 256, 0, stream>>>(embp, F(27), F(28), ss_ca, BB, TE_, 2 * DD);
  emb_ss_kernel<<<(BB * 2 * DD + 255) / 256, 256, 0, stream>>>(embp, F(37), F(38), ss_ff, BB, TE_, 2 * DD);

  dim3 gBT((unsigned)(BT / 64), DD / 128);
  dim3 gBN((unsigned)(BN / 64), DD / 128);
  dim3 gL1((unsigned)(BT / 64), FF / 128);
  unsigned attnGrid = (unsigned)(BB * HH * (TT / 64));

  // ---- Temporal self-attention ----
  ln_kernel<<<(unsigned)BT, 256, 0, stream>>>(x, F(3), F(4), xn, DD);
  gemm_wmma<<<gBT, 256, 0, stream>>>(xn, wq_sa, F(6), nullptr, nullptr, qb, (int)BT, DD, DD, 0);
  gemm_wmma<<<gBT, 256, 0, stream>>>(xn, wk_sa, F(8), nullptr, nullptr, kb, (int)BT, DD, DD, 0);
  gemm_wmma<<<gBT, 256, 0, stream>>>(xn, wv_sa, F(10), nullptr, nullptr, vb, (int)BT, DD, DD, 0);
  attn_kernel<<<attnGrid, 128, 0, stream>>>(qb, kb, vb, yb, TT, TT);
  styl_mod_kernel<<<(unsigned)BT, 256, 0, stream>>>(yb, F(13), F(14), ss_sa, hb, DD, TT);
  gemm_wmma<<<gBT, 256, 0, stream>>>(hb, wo_sa, F(16), x, x1, nullptr, (int)BT, DD, DD, 0);

  // ---- Temporal cross-attention ----
  ln_kernel<<<(unsigned)BT, 256, 0, stream>>>(x1, F(17), F(18), xn, DD);
  ln_kernel<<<(unsigned)BN, 256, 0, stream>>>(xf, F(19), F(20), cn, DD);
  gemm_wmma<<<gBT, 256, 0, stream>>>(xn, wq_ca, F(22), nullptr, nullptr, qb, (int)BT, DD, DD, 0);
  gemm_wmma<<<gBN, 256, 0, stream>>>(cn, wk_ca, F(24), nullptr, nullptr, kb, (int)BN, DD, DD, 0);
  gemm_wmma<<<gBN, 256, 0, stream>>>(cn, wv_ca, F(26), nullptr, nullptr, vb, (int)BN, DD, DD, 0);
  attn_kernel<<<attnGrid, 128, 0, stream>>>(qb, kb, vb, yb, TT, NC);
  styl_mod_kernel<<<(unsigned)BT, 256, 0, stream>>>(yb, F(29), F(30), ss_ca, hb, DD, TT);
  gemm_wmma<<<gBT, 256, 0, stream>>>(hb, wo_ca, F(32), x1, x2, x2b, (int)BT, DD, DD, 0);

  // ---- FFN ----
  gemm_wmma<<<gL1, 256, 0, stream>>>(x2b, wl1, F(34), nullptr, nullptr, h1, (int)BT, FF, DD, 2);
  gemm_wmma<<<gBT, 256, 0, stream>>>(h1, wl2, F(36), nullptr, yb, nullptr, (int)BT, DD, FF, 0);
  styl_mod_kernel<<<(unsigned)BT, 256, 0, stream>>>(yb, F(39), F(40), ss_ff, hb, DD, TT);
  gemm_wmma<<<gBT, 256, 0, stream>>>(hb, wo_ff, F(42), x2, (float*)d_out, nullptr, (int)BT, DD, DD, 0);
}
